// TrajGRUCell_12515534700746
// MI455X (gfx1250) — matrix operand order repaired and checked by reading, not compiled, not run on hardware
//
#include <hip/hip_runtime.h>
#include <math.h>

// ---------------------------------------------------------------------------
// TrajGRU cell for MI455X (gfx1250, wave32). All conv/GEMM work goes through
// v_wmma_f32_16x16x32_bf16 (bf16 operands, f32 accumulation). Activations are
// NHWC, weights channel-minor K (k = s*C + c, C a power of two). Conv K-loops
// are fully unrolled over compile-time taps so A/B addresses are base +
// immediate offsets and the only per-tap runtime work is a range check.
// The warped tensor never touches HBM: 10.5KB LDS tile per 16-pixel block,
// laid out [n][k'] so its B-fragments are ds_load_b128 pairs.
// ---------------------------------------------------------------------------

typedef __attribute__((ext_vector_type(16))) __bf16 v16bf;
typedef __attribute__((ext_vector_type(8)))  __bf16 v8bf;
typedef __attribute__((ext_vector_type(8)))  float  v8f;

#define DEVINL static __device__ __forceinline__

DEVINL v8f wmma_bf16(v16bf a, v16bf b, v8f c) {
  // (neg_a, A, neg_b, B, c_mod, C, reuse_a, reuse_b)
  return __builtin_amdgcn_wmma_f32_16x16x32_bf16(false, a, false, b,
                                                 (short)0, c, false, false);
}

DEVINL __bf16 f2bf(float f) { return (__bf16)f; }
DEVINL int imin(int a, int b) { return a < b ? a : b; }

// A fragment: bf16 weights W[M][ldk] (zero padded), rows m0..m0+15, k in
// [kb,kb+32). lane -> M, element e -> K =
// e + 16*(lane>=16)  => two aligned 16B loads at +0 and +8 elems.
DEVINL v16bf load_a(const __bf16* __restrict__ W, int ldk, int m0, int kb,
                    int lane) {
  const __bf16* p = W + (m0 + (lane & 15)) * ldk + kb + (lane >> 4) * 16;
  union { v16bf v; v8bf h[2]; } u;
  u.h[0] = *(const v8bf*)(p);
  u.h[1] = *(const v8bf*)(p + 8);
  return u.v;
}

// B fragment for a single compile-time conv tap (C >= 16): 16 consecutive
// channels (cb + half*16 ..) of NHWC pixel (y+ky-PAD, nx+kx-PAD), zero if the
// tap falls outside the image. Exec-masked loads keep addresses as
// base + literal offsets.
template <int C, int PAD>
DEVINL v16bf load_b_tap16(const __bf16* __restrict__ img, int y, int ky,
                          int nx, int kx, int cb, int lane) {
  const int yy = y + ky - PAD;
  const int xx = nx + kx - PAD;
  union { v16bf v; v8bf h[2]; } u;
  const v8bf z = {};
  u.h[0] = z;
  u.h[1] = z;
  if (((unsigned)yy < 128u) & ((unsigned)xx < 128u)) {
    const __bf16* p = img + ((yy * 128 + xx) * C + cb + (lane >> 4) * 16);
    u.h[0] = *(const v8bf*)(p);
    u.h[1] = *(const v8bf*)(p + 8);
  }
  return u.v;
}

// Generic B fragment (used for C==8 segments where one fragment spans two
// taps that depend on the lane half): k = s*C + c, s = ky*KW + kx.
template <int C, int KW, int PAD, int KHW>
DEVINL v16bf load_b_nhwc(const __bf16* __restrict__ img, int y, int nx, int kb,
                         int lane) {
  constexpr int EPG = (C < 16) ? C : 16;
  constexpr int NG = 16 / EPG;
  const int k0 = kb + (lane >> 4) * 16;
  union { v16bf v; v8bf h[2]; } u;
  const v8bf z = {};
#pragma unroll
  for (int g = 0; g < NG; ++g) {
    const int k = k0 + g * EPG;
    const int s = k / C;  // shift
    const int c0 = k & (C - 1);
    const int ky = s / KW;
    const int kx = s - ky * KW;
    const int yy = y + ky - PAD;
    const int xx = nx + kx - PAD;
    const bool inb = (s < KHW) & ((unsigned)yy < 128u) & ((unsigned)xx < 128u);
    v8bf a = z, b = z;
    if (inb) {
      const __bf16* p = img + ((yy * 128 + xx) * C + c0);
      a = *(const v8bf*)(p);
      if (EPG == 16) b = *(const v8bf*)(p + 8);
    }
    if (EPG == 16) {
      u.h[0] = a;
      u.h[1] = b;
    } else {
      u.h[g] = a;
    }
  }
  return u.v;
}

// B fragment from the warped LDS tile, layout sW[n][k'] with row stride 328.
DEVINL v16bf load_b_warp(const __bf16* __restrict__ sw, int kb, int lane) {
  const __bf16* p = sw + (lane & 15) * 328 + kb + (lane >> 4) * 16;
  union { v16bf v; v8bf h[2]; } u;
  u.h[0] = *(const v8bf*)(p);
  u.h[1] = *(const v8bf*)(p + 8);
  return u.v;
}

// ---------------------------------------------------------------------------
// prep kernels
// ---------------------------------------------------------------------------
__global__ void k_nhwc_bf16(const float* __restrict__ src,
                            __bf16* __restrict__ dst, int C, int HW,
                            int total) {
  for (int i = blockIdx.x * blockDim.x + threadIdx.x; i < total;
       i += gridDim.x * blockDim.x) {
    int c = i % C, t = i / C;
    int pix = t % HW, b = t / HW;
    dst[i] = f2bf(src[(b * C + c) * HW + pix]);
  }
}

__global__ void k_nhwc_f32(const float* __restrict__ src,
                           float* __restrict__ dst, int C, int HW, int total) {
  for (int i = blockIdx.x * blockDim.x + threadIdx.x; i < total;
       i += gridDim.x * blockDim.x) {
    int c = i % C, t = i / C;
    int pix = t % HW, b = t / HW;
    dst[i] = src[(b * C + c) * HW + pix];
  }
}

// weights [Msrc][C][KH][KW] f32 -> bf16 [Mdst][ldk], k = s*C + c, zero padded
__global__ void k_pack_w(const float* __restrict__ src,
                         __bf16* __restrict__ dst, int Msrc, int Mdst, int C,
                         int KHW, int ldk) {
  int total = Mdst * ldk;
  for (int i = blockIdx.x * blockDim.x + threadIdx.x; i < total;
       i += gridDim.x * blockDim.x) {
    int m = i / ldk, k = i - m * ldk;
    int s = k / C, c = k - s * C;
    float v = (m < Msrc && s < KHW) ? src[(m * C + c) * KHW + s] : 0.f;
    dst[i] = f2bf(v);
  }
}

// ---------------------------------------------------------------------------
// stage 1: f = tanh(conv5x5(x;Wi2f)+b_i2f + conv5x5(h;Wh2f)+b_h2f), NHWC out
// One wave per 16-pixel tile; two 16-row M-tiles (Cout=32) share B fragments.
// ---------------------------------------------------------------------------
__global__ void __launch_bounds__(128)
k_flowfeat(const __bf16* __restrict__ xb, const __bf16* __restrict__ hb,
           const __bf16* __restrict__ Wi2f, const float* __restrict__ bi2f,
           const __bf16* __restrict__ Wh2f, const float* __restrict__ bh2f,
           __bf16* __restrict__ fb) {
  const int lane = threadIdx.x & 31;
  const int tile = blockIdx.x * 4 + (threadIdx.x >> 5);  // 8192 tiles
  const int tx = tile & 7, y = (tile >> 3) & 127, b = tile >> 10;
  const int x0 = tx * 16, nx = x0 + (lane & 15);
  const __bf16* xB = xb + b * (128 * 128 * 8);
  const __bf16* hB = hb + b * (128 * 128 * 64);
  v8f c0 = {}, c1 = {};
#pragma unroll
  for (int kb = 0; kb < 224; kb += 32) {  // x seg: K = 25*8 = 200, padded
    v16bf bf = load_b_nhwc<8, 5, 2, 25>(xB, y, nx, kb, lane);
    c0 = wmma_bf16(load_a(Wi2f, 224, 0, kb, lane), bf, c0);
    c1 = wmma_bf16(load_a(Wi2f, 224, 16, kb, lane), bf, c1);
  }
  // h seg: K = 25*64, compile-time taps
#pragma unroll
  for (int ky = 0; ky < 5; ++ky)
#pragma unroll
    for (int kx = 0; kx < 5; ++kx)
#pragma unroll
      for (int cb = 0; cb < 64; cb += 32) {
        const int kb = (ky * 5 + kx) * 64 + cb;
        v16bf bf = load_b_tap16<64, 2>(hB, y, ky, nx, kx, cb, lane);
        c0 = wmma_bf16(load_a(Wh2f, 1600, 0, kb, lane), bf, c0);
        c1 = wmma_bf16(load_a(Wh2f, 1600, 16, kb, lane), bf, c1);
      }
  const int n = lane & 15, half = lane >> 4;
  v8bf o0, o1;
#pragma unroll
  for (int r = 0; r < 8; ++r) {
    int m = r + half * 8;
    o0[r] = f2bf(tanhf(c0[r] + bi2f[m] + bh2f[m]));
    o1[r] = f2bf(tanhf(c1[r] + bi2f[m + 16] + bh2f[m + 16]));
  }
  __bf16* fB = fb + b * (128 * 128 * 32) + (y * 128 + x0 + n) * 32;
  *(v8bf*)(fB + half * 8) = o0;
  *(v8bf*)(fB + 16 + half * 8) = o1;
}

// ---------------------------------------------------------------------------
// stage 2: flows = conv5x5(f;Wflows) + b_flows   (M = 10 padded to 16)
// ---------------------------------------------------------------------------
__global__ void __launch_bounds__(128)
k_flows(const __bf16* __restrict__ fb, const __bf16* __restrict__ Wfl,
        const float* __restrict__ bfl, float* __restrict__ flows) {
  const int lane = threadIdx.x & 31;
  const int tile = blockIdx.x * 4 + (threadIdx.x >> 5);
  const int tx = tile & 7, y = (tile >> 3) & 127, b = tile >> 10;
  const int x0 = tx * 16, nx = x0 + (lane & 15);
  const __bf16* fB = fb + b * (128 * 128 * 32);
  v8f c = {};
#pragma unroll
  for (int ky = 0; ky < 5; ++ky)
#pragma unroll
    for (int kx = 0; kx < 5; ++kx) {
      const int kb = (ky * 5 + kx) * 32;  // K = 25*32, one k-step per tap
      v16bf bf = load_b_tap16<32, 2>(fB, y, ky, nx, kx, 0, lane);
      c = wmma_bf16(load_a(Wfl, 800, 0, kb, lane), bf, c);
    }
  const int n = lane & 15, half = lane >> 4;
#pragma unroll
  for (int r = 0; r < 8; ++r) {
    int m = r + half * 8;
    if (m < 10)
      flows[((b * 10 + m) * 128 + y) * 128 + x0 + n] = c[r] + bfl[m];
  }
}

// ---------------------------------------------------------------------------
// stage 3 (fused): per 16-pixel tile
//   - warped tile [16 n][320 k'] bf16 in LDS (bilinear, border clamp), built
//     from the NHWC f32 h_prev copy (lanes sweep channels -> coalesced)
//   - GEMMs: Cr/Cu fuse i2h(K=96pad)+h2h(K=320); Cci = i2h only; Cch = h2h
//     only (rows c0 / c0+64 / c0+128 of the 192-row weight matrices)
//   - GRU gating epilogue, f32 NCHW output
// 4 waves/block: wave w -> output channels [16w, 16w+16).
// ---------------------------------------------------------------------------
__global__ void __launch_bounds__(128)
k_gru_out(const float* __restrict__ hprev, const float* __restrict__ hnf,
          const __bf16* __restrict__ xb, const float* __restrict__ flows,
          const __bf16* __restrict__ Wi2h, const float* __restrict__ bi2h,
          const __bf16* __restrict__ Wret, const float* __restrict__ bret,
          float* __restrict__ out) {
  __shared__ __bf16 sW[16 * 328];  // [n][k'], padded rows (10.5 KB)
  const int tile = blockIdx.x;
  const int tx = tile & 7, y = (tile >> 3) & 127, b = tile >> 10;
  const int x0 = tx * 16;
  const float* hN = hnf + b * (128 * 128 * 64);

  // cooperative warp: sample h_prev at (x - fx, y - fy), border-clamped.
  for (int idx = threadIdx.x; idx < 5120; idx += 128) {
    int c = idx & 63, n = (idx >> 6) & 15, l = idx >> 10;
    float fx = flows[((b * 10 + 2 * l) * 128 + y) * 128 + x0 + n];
    float fy = flows[((b * 10 + 2 * l + 1) * 128 + y) * 128 + x0 + n];
    float px = fminf(fmaxf((float)(x0 + n) - fx, 0.f), 127.f);
    float py = fminf(fmaxf((float)y - fy, 0.f), 127.f);
    int xi = (int)floorf(px), yi = (int)floorf(py);
    int x1 = imin(xi + 1, 127), y1 = imin(yi + 1, 127);
    float wx = px - (float)xi, wy = py - (float)yi;
    float v = hN[(yi * 128 + xi) * 64 + c] * (1.f - wx) * (1.f - wy) +
              hN[(yi * 128 + x1) * 64 + c] * wx * (1.f - wy) +
              hN[(y1 * 128 + xi) * 64 + c] * (1.f - wx) * wy +
              hN[(y1 * 128 + x1) * 64 + c] * wx * wy;
    sW[n * 328 + l * 64 + c] = f2bf(v);
  }
  __syncthreads();

  const int lane = threadIdx.x & 31, wave = threadIdx.x >> 5;
  const int c0 = wave * 16;
  const int nx = x0 + (lane & 15);
  const __bf16* xB = xb + b * (128 * 128 * 8);
  v8f Cr = {}, Cu = {}, Cci = {}, Cch = {};
#pragma unroll
  for (int kb = 0; kb < 96; kb += 32) {  // i2h: K = 9*8 = 72, padded
    v16bf bf = load_b_nhwc<8, 3, 1, 9>(xB, y, nx, kb, lane);
    Cr  = wmma_bf16(load_a(Wi2h, 96, c0, kb, lane), bf, Cr);
    Cu  = wmma_bf16(load_a(Wi2h, 96, c0 + 64, kb, lane), bf, Cu);
    Cci = wmma_bf16(load_a(Wi2h, 96, c0 + 128, kb, lane), bf, Cci);
  }
#pragma unroll
  for (int kb = 0; kb < 320; kb += 32) {  // h2h over LDS warped tile
    v16bf bf = load_b_warp(sW, kb, lane);
    Cr  = wmma_bf16(load_a(Wret, 320, c0, kb, lane), bf, Cr);
    Cu  = wmma_bf16(load_a(Wret, 320, c0 + 64, kb, lane), bf, Cu);
    Cch = wmma_bf16(load_a(Wret, 320, c0 + 128, kb, lane), bf, Cch);
  }
  const int n = lane & 15, half = lane >> 4;
  const float* hB = hprev + b * (64 * 128 * 128);
#pragma unroll
  for (int r = 0; r < 8; ++r) {
    int c = c0 + r + half * 8;
    float reset = 1.f / (1.f + expf(-(Cr[r] + bi2h[c] + bret[c])));
    float upd   = 1.f / (1.f + expf(-(Cu[r] + bi2h[c + 64] + bret[c + 64])));
    float cand  = tanhf(Cci[r] + bi2h[c + 128] +
                        reset * (Cch[r] + bret[c + 128]));
    int o = (c * 128 + y) * 128 + x0 + n;
    out[b * (64 * 128 * 128) + o] = upd * hB[o] + (1.f - upd) * cand;
  }
}

// ---------------------------------------------------------------------------
extern "C" void kernel_launch(void* const* d_in, const int* in_sizes, int n_in,
                              void* d_out, int out_size, void* d_ws,
                              size_t ws_size, hipStream_t stream) {
  const float* x     = (const float*)d_in[0];
  const float* hprev = (const float*)d_in[1];
  const float* w_i2f = (const float*)d_in[2];
  const float* b_i2f = (const float*)d_in[3];
  const float* w_h2f = (const float*)d_in[4];
  const float* b_h2f = (const float*)d_in[5];
  const float* w_fl  = (const float*)d_in[6];
  const float* b_fl  = (const float*)d_in[7];
  const float* w_i2h = (const float*)d_in[8];
  const float* b_i2h = (const float*)d_in[9];
  const float* w_ret = (const float*)d_in[10];
  const float* b_ret = (const float*)d_in[11];
  float* out = (float*)d_out;

  char* ws = (char*)d_ws;
  size_t off = 0;
  auto carve = [&](size_t bytes) {
    void* p = ws + off;
    off = (off + bytes + 255) & ~(size_t)255;
    return p;
  };
  __bf16* Wi2f = (__bf16*)carve(32 * 224 * 2);
  __bf16* Wh2f = (__bf16*)carve(32 * 1600 * 2);
  __bf16* Wfl  = (__bf16*)carve(16 * 800 * 2);
  __bf16* Wi2h = (__bf16*)carve(192 * 96 * 2);
  __bf16* Wret = (__bf16*)carve(192 * 320 * 2);
  __bf16* xb   = (__bf16*)carve((size_t)8 * 128 * 128 * 8 * 2);   // NHWC
  __bf16* hb   = (__bf16*)carve((size_t)8 * 128 * 128 * 64 * 2);  // NHWC
  __bf16* fb   = (__bf16*)carve((size_t)8 * 128 * 128 * 32 * 2);  // NHWC
  float*  hnf  = (float*) carve((size_t)8 * 128 * 128 * 64 * 4);  // NHWC f32
  float*  fl   = (float*) carve((size_t)8 * 10 * 128 * 128 * 4);  // NCHW

  // weight packing (bf16, channel-minor K, zero padded)
  k_pack_w<<<(32 * 224 + 255) / 256, 256, 0, stream>>>(w_i2f, Wi2f, 32, 32, 8, 25, 224);
  k_pack_w<<<(32 * 1600 + 255) / 256, 256, 0, stream>>>(w_h2f, Wh2f, 32, 32, 64, 25, 1600);
  k_pack_w<<<(16 * 800 + 255) / 256, 256, 0, stream>>>(w_fl, Wfl, 10, 16, 32, 25, 800);
  k_pack_w<<<(192 * 96 + 255) / 256, 256, 0, stream>>>(w_i2h, Wi2h, 192, 192, 8, 9, 96);
  k_pack_w<<<(192 * 320 + 255) / 256, 256, 0, stream>>>(w_ret, Wret, 192, 192, 320, 1, 320);

  // activation layout conversion
  k_nhwc_bf16<<<2048, 256, 0, stream>>>(x, xb, 8, 128 * 128, 8 * 8 * 128 * 128);
  k_nhwc_bf16<<<8192, 256, 0, stream>>>(hprev, hb, 64, 128 * 128, 8 * 64 * 128 * 128);
  k_nhwc_f32<<<8192, 256, 0, stream>>>(hprev, hnf, 64, 128 * 128, 8 * 64 * 128 * 128);

  // pipeline (8192 16-pixel tiles; 4 wave-tiles/block for stages 1-2)
  k_flowfeat<<<2048, 128, 0, stream>>>(xb, hb, Wi2f, b_i2f, Wh2f, b_h2f, fb);
  k_flows<<<2048, 128, 0, stream>>>(fb, Wfl, b_fl, fl);
  k_gru_out<<<8192, 128, 0, stream>>>(hprev, hnf, xb, fl, Wi2h, b_i2h, Wret,
                                      b_ret, out);
}
